// HoughSplitLoss_12884901888574
// MI455X (gfx1250) — compile-verified, matrix operand order checked by reading, not run on hardware
//
#include <hip/hip_runtime.h>

typedef __attribute__((ext_vector_type(16))) _Float16 v16h;
typedef __attribute__((ext_vector_type(8)))  float    v8f;

#define RADIUS   5
#define WIN      (2 * RADIUS + 1)   // 11
#define WIN2     (WIN * WIN)        // 121
#define REG_TH   10.0f
#define BIG_KEY  (1 << 30)
#define ADIM     1024               // angle dim (A)
#define RDIM     1024               // radius dim (R)
#define LPTS     6                  // points per batch

// workspace layout (floats)
#define WS_TOTAL 0                  // ws[0] = total
#define WS_PT    8                  // ws[8 .. 8+B*L-1] per-point losses
#define WS_FB    256                // ws[256 .. 256+NBLK_FB-1] fallback partials
#define NBLK_FB  1024
#define THR_FB   256

__device__ __forceinline__ float sigmoid_fast(float x) {
    // v_rcp_f32 instead of IEEE div: avoids the DIV_SCALE/DIV_FMAS/DIV_FIXUP
    // chain (which crashed ISel when SLP-vectorized) and is faster.
    return __builtin_amdgcn_rcpf(1.0f + __expf(-x));
}

// |x| as a single VALU op, opaque to the SLP vectorizer. The compiler's fabs
// on SLP-paired v2f32 values hits an ISel bug (malformed bitcast/and node),
// so we emit v_max_f32 r, x, -x (neg source modifier) via inline asm.
__device__ __forceinline__ float fabs_opaque(float x) {
    float r;
    asm("v_max_f32_e64 %0, %1, -%1" : "=v"(r) : "v"(x));
    return r;
}

// ---------------------------------------------------------------------------
// K1: one block per (batch, point-slot). 121 lanes gather the 11x11 window.
// ---------------------------------------------------------------------------
__global__ void __launch_bounds__(128)
point_loss_kernel(const float* __restrict__ pre, const float* __restrict__ gth,
                  const int* __restrict__ prept, const int* __restrict__ gtpt,
                  float* __restrict__ ws) {
    const int bl = blockIdx.x;
    const int b  = bl / LPTS;
    const int l  = bl % LPTS;
    const int t  = threadIdx.x;

    // Load this batch's 6 pre-points; build stable-sort keys.
    int px[LPTS], py[LPTS], key[LPTS];
#pragma unroll
    for (int j = 0; j < LPTS; ++j) {
        px[j]  = prept[(b * LPTS + j) * 2 + 0];
        py[j]  = prept[(b * LPTS + j) * 2 + 1];
        key[j] = (px[j] < 0) ? BIG_KEY : (px[j] * 10000 + py[j]);
    }
    // Stable rank-select: element whose rank == l is the l-th sorted pre-point.
    int selx = 0, sely = 0;
#pragma unroll
    for (int j = 0; j < LPTS; ++j) {
        int rank = 0;
#pragma unroll
        for (int i = 0; i < LPTS; ++i)
            rank += (key[i] < key[j]) || (key[i] == key[j] && i < j);
        if (rank == l) { selx = px[j]; sely = py[j]; }
    }

    const int gx = gtpt[(b * LPTS + l) * 2 + 0];
    const int gy = gtpt[(b * LPTS + l) * 2 + 1];
    const bool use_gt = (gx >= 0);
    const int x = use_gt ? gx : selx;
    const int y = use_gt ? gy : sely;

    float v = 0.0f;
    if (t < WIN2) {
        const int dx = t / WIN - RADIUS;
        const int dy = t % WIN - RADIUS;
        const int xi = x + dx;
        const int yi = y + dy;
        if (xi >= 0 && xi < RDIM && yi >= 0 && yi < ADIM) {
            const size_t idx = (size_t)b * ADIM * RDIM + (size_t)xi * RDIM + yi;
            const float s = sigmoid_fast(pre[idx]);
            v = fabs_opaque(s - gth[idx]);
        }
    }

    __shared__ float red[128];
    red[t] = v;
    __syncthreads();
    for (int s = 64; s > 0; s >>= 1) {
        if (t < s) red[t] += red[t + s];
        __syncthreads();
    }

    if (t == 0) {
        const bool pre_inv = (selx < 0);
        const bool gt_inv  = (gx < 0);
        float out;
        if (pre_inv && gt_inv) out = 0.0f;
        else out = red[0] * ((pre_inv || gt_inv) ? REG_TH : 1.0f)
                          * (1.0f / (float)(RADIUS * RADIUS));
        ws[WS_PT + bl] = out;
    }
}

// ---------------------------------------------------------------------------
// K2: reduce per-point losses -> ws[0] = total (fixed-order, deterministic)
// ---------------------------------------------------------------------------
__global__ void __launch_bounds__(256)
total_reduce_kernel(float* __restrict__ ws, int npts) {
    const int t = threadIdx.x;
    __shared__ float red[256];
    red[t] = (t < npts) ? ws[WS_PT + t] : 0.0f;
    __syncthreads();
    for (int s = 128; s > 0; s >>= 1) {
        if (t < s) red[t] += red[t + s];
        __syncthreads();
    }
    if (t == 0) ws[WS_TOTAL] = red[0];
}

// ---------------------------------------------------------------------------
// K3: fallback partial sums of |sigmoid(pre) - gt| over all N elements.
//     Early-exits if total != 0 (the common case) -> skips 268 MB of traffic.
//     Accumulation runs on the matrix pipe: 16 f16 |diff| values per lane per
//     step, v_wmma_f32_16x16x32_f16 against all-ones B; f32 C accumulates.
//     Each element's row-sum lands in 16 columns -> scale by 1/16 at the end.
// ---------------------------------------------------------------------------
__global__ void __launch_bounds__(THR_FB)
fallback_partial_kernel(const float* __restrict__ pre,
                        const float* __restrict__ gth,
                        float* __restrict__ ws, size_t N) {
    const float total = ws[WS_TOTAL];
    if (total != 0.0f) return;   // uniform: taken by every thread or none

    const size_t gtid     = (size_t)blockIdx.x * blockDim.x + threadIdx.x;
    const size_t nthreads = (size_t)gridDim.x * blockDim.x;
    const size_t N4       = N >> 2;              // N is a multiple of 4 here
    const float4* a4 = (const float4*)pre;
    const float4* g4 = (const float4*)gth;

    v16h bones;
#pragma unroll
    for (int i = 0; i < 16; ++i) bones[i] = (_Float16)1.0f;

    v8f  c = {};
    float tail = 0.0f;

    // Wave-uniform trip count so EXEC is all-ones at every WMMA.
    const size_t stride = 4 * nthreads;          // 4 float4 (16 elems)/lane/step
    const size_t chunks = N4 / stride;

    for (size_t k = 0; k < chunks; ++k) {
        const size_t base = gtid + k * stride;
        __builtin_prefetch(a4 + base + stride, 0, 1);   // global_prefetch_b8
        __builtin_prefetch(g4 + base + stride, 0, 1);
        v16h a;
#pragma unroll
        for (int u = 0; u < 4; ++u) {
            const float4 va = a4[base + (size_t)u * nthreads];
            const float4 vg = g4[base + (size_t)u * nthreads];
            a[4 * u + 0] = (_Float16)fabs_opaque(sigmoid_fast(va.x) - vg.x);
            a[4 * u + 1] = (_Float16)fabs_opaque(sigmoid_fast(va.y) - vg.y);
            a[4 * u + 2] = (_Float16)fabs_opaque(sigmoid_fast(va.z) - vg.z);
            a[4 * u + 3] = (_Float16)fabs_opaque(sigmoid_fast(va.w) - vg.w);
        }
        // D = A x ones + C  (row sums accumulate into all 16 columns of C)
        c = __builtin_amdgcn_wmma_f32_16x16x32_f16(
                /*neg_a=*/false, a, /*neg_b=*/false, bones,
                /*c_mod=*/(short)0, c, /*reuse_a=*/false, /*reuse_b=*/false);
    }
    // Remainder (none for these shapes), plain f32 accumulation.
    for (size_t i4 = gtid + chunks * stride; i4 < N4; i4 += nthreads) {
        const float4 va = a4[i4];
        const float4 vg = g4[i4];
        tail += fabs_opaque(sigmoid_fast(va.x) - vg.x);
        tail += fabs_opaque(sigmoid_fast(va.y) - vg.y);
        tail += fabs_opaque(sigmoid_fast(va.z) - vg.z);
        tail += fabs_opaque(sigmoid_fast(va.w) - vg.w);
    }

    float lane = (c[0] + c[1] + c[2] + c[3] + c[4] + c[5] + c[6] + c[7])
                 * (1.0f / 16.0f) + tail;

    __shared__ float red[THR_FB];
    const int t = threadIdx.x;
    red[t] = lane;
    __syncthreads();
    for (int s = THR_FB / 2; s > 0; s >>= 1) {
        if (t < s) red[t] += red[t + s];
        __syncthreads();
    }
    if (t == 0) ws[WS_FB + blockIdx.x] = red[0];
}

// ---------------------------------------------------------------------------
// K4: finalize. total != 0 -> total/B ; else reduce partials -> mean.
// ---------------------------------------------------------------------------
__global__ void __launch_bounds__(256)
finalize_kernel(const float* __restrict__ ws, float* __restrict__ out,
                float Bf, size_t N) {
    const float total = ws[WS_TOTAL];
    const int t = threadIdx.x;
    if (total != 0.0f) {
        if (t == 0) out[0] = total / Bf;
        return;
    }
    __shared__ float red[256];
    float v = 0.0f;
#pragma unroll
    for (int u = 0; u < NBLK_FB / 256; ++u)
        v += ws[WS_FB + t + u * 256];
    red[t] = v;
    __syncthreads();
    for (int s = 128; s > 0; s >>= 1) {
        if (t < s) red[t] += red[t + s];
        __syncthreads();
    }
    if (t == 0) out[0] = red[0] / (float)N;
}

// ---------------------------------------------------------------------------
extern "C" void kernel_launch(void* const* d_in, const int* in_sizes, int n_in,
                              void* d_out, int out_size, void* d_ws, size_t ws_size,
                              hipStream_t stream) {
    const float* pre   = (const float*)d_in[0];   // pre_hough  (B,1,A,R) f32
    const float* gth   = (const float*)d_in[1];   // gt_hough   (B,1,A,R) f32
    const int*   prept = (const int*)d_in[2];     // in_pre_points (B,L,2) i32
    const int*   gtpt  = (const int*)d_in[3];     // gt_points     (B,L,2) i32
    float*       out   = (float*)d_out;
    float*       ws    = (float*)d_ws;

    const int    B = in_sizes[2] / (2 * LPTS);    // 32
    const size_t N = (size_t)in_sizes[0];         // B*A*R = 33,554,432

    point_loss_kernel<<<B * LPTS, 128, 0, stream>>>(pre, gth, prept, gtpt, ws);
    total_reduce_kernel<<<1, 256, 0, stream>>>(ws, B * LPTS);
    fallback_partial_kernel<<<NBLK_FB, THR_FB, 0, stream>>>(pre, gth, ws, N);
    finalize_kernel<<<1, 256, 0, stream>>>(ws, out, (float)B, N);
}
